// Classifer_22299470201682
// MI455X (gfx1250) — matrix operand order repaired and checked by reading, not compiled, not run on hardware
//
#include <hip/hip_runtime.h>
#include <hip/hip_bf16.h>
#include <math.h>

typedef __attribute__((ext_vector_type(16))) _Float16 v16h;
typedef __attribute__((ext_vector_type(8)))  float    v8f;

#define CDIV(a, b) (((a) + (b) - 1) / (b))

static constexpr int   NN    = 100000;
static constexpr int   EE    = 1600000;
static constexpr int   GG    = 1024;
static constexpr int   HH    = 128;
static constexpr int   F_IN  = 38;
static constexpr float EPS   = 1e-5f;

// ---------------------------------------------------------------- degrees
__global__ void k_deg(const int* __restrict__ src, const int* __restrict__ dst,
                      float* degO, float* degI, int E) {
    int i = blockIdx.x * blockDim.x + threadIdx.x;
    if (i < E) {
        atomicAdd(&degO[src[i]], 1.0f);
        atomicAdd(&degI[dst[i]], 1.0f);
    }
}

__global__ void k_norm(const float* __restrict__ degO, const float* __restrict__ degI,
                       float* ns, float* nd, int N) {
    int i = blockIdx.x * blockDim.x + threadIdx.x;
    if (i < N) {
        ns[i] = rsqrtf(fmaxf(degO[i], 1.0f));
        nd[i] = rsqrtf(fmaxf(degI[i], 1.0f));
    }
}

// ------------------------------------------------- edge gather/scatter-add
// one block per edge, thread f = feature; msg = x[src]*ns[src] -> agg[dst]
__global__ void k_scatter(const float* __restrict__ X, const int* __restrict__ src,
                          const int* __restrict__ dst, const float* __restrict__ ns,
                          float* agg, int F) {
    int e = blockIdx.x;
    int f = threadIdx.x;
    int s = src[e];
    int d = dst[e];
    float v = X[(size_t)s * F + f] * ns[s];
    atomicAdd(&agg[(size_t)d * F + f], v);
}

// -------------------------------------------------- fragment-packed staging
// A fragments are stored pre-swizzled into the exact WMMA VGPR layout:
//   Apk[((tileM*KC + kc)*32 + lane)*16 + i]  == half i of lane's A frag
// 16-bit A 16x32 layout: lanes 0-15 M rows, VGPR0-3 -> K 0..7, VGPR4-7 -> K 16..23,
// lanes 16-31 K offset +8.  Optional per-row scale (nd), K zero-padded to Kp.
__global__ void k_packA(const float* __restrict__ S, const float* __restrict__ scale,
                        _Float16* Apk, int M, int F, int Kp) {
    size_t idx = (size_t)blockIdx.x * blockDim.x + threadIdx.x;
    if (idx >= (size_t)M * Kp) return;
    const int KC   = Kp >> 5;
    const int i    = (int)(idx & 15);
    const int lane = (int)((idx >> 4) & 31);
    const size_t rest = idx >> 9;
    const int kc = (int)(rest % KC);
    const int tm = (int)(rest / KC);
    const int row  = tm * 16 + (lane & 15);
    const int pair = i >> 1;
    const int half = i & 1;
    const int k = kc * 32 + (lane >> 4) * 8 +
                  ((pair < 4) ? 2 * pair : 16 + 2 * (pair - 4)) + half;
    float v = 0.0f;
    if (k < F) {
        v = S[(size_t)row * F + k];
        if (scale) v *= scale[row];
    }
    Apk[idx] = (_Float16)v;
}

// B fragments pre-swizzled:
//   Bpk[((tileN*KC + kc)*32 + lane)*16 + i]
// 16-bit B 32x16: lanes 0-15 VGPR0-3 -> K 0..7, VGPR4-7 -> K 16..23, lanes 16-31 K offset +8; 2 K per VGPR.
// Source W is row-major [rows x Nn], K zero-padded to Kp.
__global__ void k_packB(const float* __restrict__ W, _Float16* Bpk,
                        int rows, int Kp, int Nn) {
    size_t idx = (size_t)blockIdx.x * blockDim.x + threadIdx.x;
    if (idx >= (size_t)Kp * Nn) return;
    const int KC   = Kp >> 5;
    const int i    = (int)(idx & 15);
    const int lane = (int)((idx >> 4) & 31);
    const size_t rest = idx >> 9;
    const int kc = (int)(rest % KC);
    const int tn = (int)(rest / KC);
    const int col  = tn * 16 + (lane & 15);
    const int pair = i >> 1;
    const int half = i & 1;
    const int k = kc * 32 + (lane >> 4) * 8 +
                  ((pair < 4) ? 2 * pair : 16 + 2 * (pair - 4)) + half;
    float v = (k < rows) ? W[(size_t)k * Nn + col] : 0.0f;
    Bpk[idx] = (_Float16)v;
}

// --------------------------------------------------------- WMMA GEMM
// C[M,Nn] = A @ B + bias, fragments pre-packed (b128 loads per lane).
// 256-thread block = 8 waves; wave w owns N-tile blockIdx.x*8+w.
// Each wave computes MT=2 M-tiles, reusing the B fragment -> KC*2 wmmas.
template <int KC>
__global__ void k_gemm_wmma(const _Float16* __restrict__ Apk,
                            const _Float16* __restrict__ Bpk,
                            const float* __restrict__ bias, float* __restrict__ C,
                            int Nn) {
    constexpr int MT = 2;
    const int lane  = threadIdx.x & 31;
    const int wid   = threadIdx.x >> 5;
    const int tileN = blockIdx.x * 8 + wid;
    const int tileM0 = blockIdx.y * MT;

    const v16h* Av = (const v16h*)Apk;
    const v16h* Bv = (const v16h*)Bpk;

    v8f acc[MT];
    #pragma unroll
    for (int m = 0; m < MT; ++m) acc[m] = (v8f){};

    const v16h* bp = Bv + ((size_t)tileN * KC) * 32 + lane;
    #pragma unroll
    for (int kc = 0; kc < KC; ++kc) {
        v16h bfrag = bp[(size_t)kc * 32];
        #pragma unroll
        for (int m = 0; m < MT; ++m) {
            v16h afrag = Av[(((size_t)(tileM0 + m) * KC) + kc) * 32 + lane];
            acc[m] = __builtin_amdgcn_wmma_f32_16x16x32_f16(
                /*neg_a=*/false, afrag, /*neg_b=*/false, bfrag,
                /*c_mod=*/(short)0, acc[m], /*reuse_a=*/false, /*reuse_b=*/false);
        }
    }

    const int lm  = lane & 15;
    const int hi  = lane >> 4;
    const int col = tileN * 16 + lm;
    const float bi = bias[col];
    #pragma unroll
    for (int m = 0; m < MT; ++m) {
        #pragma unroll
        for (int j = 0; j < 8; ++j) {
            int r = (tileM0 + m) * 16 + hi * 8 + j;   // C: VGPR j -> rows j / j+8
            C[(size_t)r * Nn + col] = acc[m][j] + bi;
        }
    }
}

// ----------------------------------------------------- BatchNorm statistics
__global__ void k_bnstats(const float* __restrict__ Z, float* csum, float* csq,
                          int M, int Nn) {
    __shared__ float s1[256];
    __shared__ float s2[256];
    const int col = blockIdx.x;
    const int tid = threadIdx.x;
    const int row = blockIdx.y * 256 + tid;
    float v = (row < M) ? Z[(size_t)row * Nn + col] : 0.0f;
    s1[tid] = v;
    s2[tid] = v * v;
    __syncthreads();
    for (int o = 128; o > 0; o >>= 1) {
        if (tid < o) { s1[tid] += s1[tid + o]; s2[tid] += s2[tid + o]; }
        __syncthreads();
    }
    if (tid == 0) {
        atomicAdd(&csum[col], s1[0]);
        atomicAdd(&csq[col],  s2[0]);
    }
}

// x = relu(g*(z-m)*rsqrt(var+eps)+be)   (in-place safe)
__global__ void k_bnapply(const float* __restrict__ Z, const float* __restrict__ csum,
                          const float* __restrict__ csq, const float* __restrict__ g,
                          const float* __restrict__ be, float* X, int M, int Nn) {
    size_t idx = (size_t)blockIdx.x * blockDim.x + threadIdx.x;
    if (idx >= (size_t)M * Nn) return;
    int col = (int)(idx % Nn);
    float invM = 1.0f / (float)M;
    float m    = csum[col] * invM;
    float var  = csq[col] * invM - m * m;
    float y    = g[col] * (Z[idx] - m) * rsqrtf(var + EPS) + be[col];
    X[idx] = fmaxf(y, 0.0f);
}

// ------------------------------------------------- gated sum readout
__global__ void k_readout(const float* __restrict__ X, const int* __restrict__ n2g,
                          const float* __restrict__ aw, const float* __restrict__ ab,
                          float* hg, float* aout) {
    __shared__ float s[128];
    const int n = blockIdx.x;
    const int f = threadIdx.x;
    float xv = X[(size_t)n * HH + f];
    s[f] = xv * aw[f];
    __syncthreads();
    for (int o = 64; o > 0; o >>= 1) {
        if (f < o) s[f] += s[f + o];
        __syncthreads();
    }
    float t = s[0] + ab[0];
    if (f == 0) aout[n] = t;
    float w = 1.0f / (1.0f + __expf(-t));
    atomicAdd(&hg[(size_t)n2g[n] * HH + f], xv * w);
}

// ------------------------------------------------- final layer + sigmoid
__global__ void k_final(const float* __restrict__ Y2, const float* __restrict__ W,
                        const float* __restrict__ b, float* __restrict__ out) {
    const int gidx = blockIdx.x;
    const int f    = threadIdx.x;
    if (f >= 67) return;
    float acc = b[f];
    const float* row = Y2 + (size_t)gidx * 256;
    #pragma unroll 4
    for (int k = 0; k < 256; ++k) acc += row[k] * W[(size_t)k * 67 + f];
    out[(size_t)gidx * 67 + f] = 1.0f / (1.0f + __expf(-acc));
}

// ======================================================================
extern "C" void kernel_launch(void* const* d_in, const int* in_sizes, int n_in,
                              void* d_out, int out_size, void* d_ws, size_t ws_size,
                              hipStream_t stream) {
    (void)in_sizes; (void)n_in; (void)out_size; (void)ws_size;

    const float* h    = (const float*)d_in[0];
    const int*   src  = (const int*)  d_in[1];
    const int*   dst  = (const int*)  d_in[2];
    const int*   n2g  = (const int*)  d_in[3];
    const float* W1   = (const float*)d_in[4];
    const float* b1   = (const float*)d_in[5];
    const float* g1   = (const float*)d_in[6];
    const float* be1  = (const float*)d_in[7];
    const float* W2s  = (const float*)d_in[8];
    const float* b2s  = (const float*)d_in[9];
    const float* g2s  = (const float*)d_in[10];
    const float* be2s = (const float*)d_in[11];
    const float* aw   = (const float*)d_in[12];
    const float* ab   = (const float*)d_in[13];
    const float* Wf1  = (const float*)d_in[14];
    const float* bf1  = (const float*)d_in[15];
    const float* gf1  = (const float*)d_in[16];
    const float* bef1 = (const float*)d_in[17];
    const float* Wl   = (const float*)d_in[18];
    const float* bl   = (const float*)d_in[19];
    const float* gl   = (const float*)d_in[20];
    const float* bel  = (const float*)d_in[21];
    const float* Wf2  = (const float*)d_in[22];
    const float* bf2  = (const float*)d_in[23];

    float* y_out  = (float*)d_out;                    // [1024, 67]
    float* aw_out = (float*)d_out + (size_t)GG * 67;  // [N, 1]

    // --- carve workspace
    char* p = (char*)d_ws;
    auto carve = [&](size_t bytes) -> char* {
        char* r = p;
        p += (bytes + 255) & ~(size_t)255;
        return r;
    };
    float*    degO = (float*)carve((size_t)NN * 4);
    float*    degI = (float*)carve((size_t)NN * 4);
    float*    ns   = (float*)carve((size_t)NN * 4);
    float*    nd   = (float*)carve((size_t)NN * 4);
    float*    x    = (float*)carve((size_t)NN * HH * 4);   // z aliased onto x
    float*    agg  = (float*)carve((size_t)NN * HH * 4);
    _Float16* Apk  = (_Float16*)carve((size_t)NN * HH * 2);
    _Float16* Bpk  = (_Float16*)carve((size_t)512 * 512 * 2);
    float*    csum = (float*)carve(512 * 4);
    float*    csq  = (float*)carve(512 * 4);
    float*    hg   = (float*)carve((size_t)GG * HH * 4);
    float*    y1   = (float*)carve((size_t)GG * 512 * 4);
    float*    y2   = (float*)carve((size_t)GG * 256 * 4);
    float*    z    = x; // alias: gemm writes z after scatter consumed x

    // --- degrees & symmetric norms
    hipMemsetAsync(degO, 0, (size_t)NN * 4, stream);
    hipMemsetAsync(degI, 0, (size_t)NN * 4, stream);
    k_deg <<<CDIV(EE, 256), 256, 0, stream>>>(src, dst, degO, degI, EE);
    k_norm<<<CDIV(NN, 256), 256, 0, stream>>>(degO, degI, ns, nd, NN);

    // --- one gconv + BN + ReLU layer (Kp = padded K, KC = Kp/32 compile-time)
    auto gconv = [&](const float* xin, int F, int Kp, const float* W, int Wrows,
                     const float* bias, const float* gam, const float* bet) {
        hipMemsetAsync(agg, 0, (size_t)NN * F * 4, stream);
        k_scatter<<<EE, F, 0, stream>>>(xin, src, dst, ns, agg, F);
        k_packA<<<CDIV((size_t)NN * Kp, 256), 256, 0, stream>>>(agg, nd, Apk, NN, F, Kp);
        k_packB<<<CDIV((size_t)Kp * HH, 256), 256, 0, stream>>>(W, Bpk, Wrows, Kp, HH);
        if (Kp == 64)
            k_gemm_wmma<2><<<dim3(HH / 128, NN / 32), 256, 0, stream>>>(Apk, Bpk, bias, z, HH);
        else
            k_gemm_wmma<4><<<dim3(HH / 128, NN / 32), 256, 0, stream>>>(Apk, Bpk, bias, z, HH);
        hipMemsetAsync(csum, 0, HH * 4, stream);
        hipMemsetAsync(csq,  0, HH * 4, stream);
        k_bnstats<<<dim3(HH, CDIV(NN, 256)), 256, 0, stream>>>(z, csum, csq, NN, HH);
        k_bnapply<<<CDIV((size_t)NN * HH, 256), 256, 0, stream>>>(z, csum, csq, gam, bet, x, NN, HH);
    };

    gconv(h, F_IN, 64, W1, F_IN, b1, g1, be1);
    for (int i = 0; i < 3; ++i)
        gconv(x, HH, HH, W2s + (size_t)i * HH * HH, HH,
              b2s + (size_t)i * HH, g2s + (size_t)i * HH, be2s + (size_t)i * HH);

    // --- WeightAndSum readout
    hipMemsetAsync(hg, 0, (size_t)GG * HH * 4, stream);
    k_readout<<<NN, 128, 0, stream>>>(x, n2g, aw, ab, hg, aw_out);

    // --- MLP head:  hg[1024,128] -> 512 -> 256 -> 67
    k_packA<<<CDIV((size_t)GG * HH, 256), 256, 0, stream>>>(hg, nullptr, Apk, GG, HH, HH);
    k_packB<<<CDIV((size_t)HH * 512, 256), 256, 0, stream>>>(Wf1, Bpk, HH, HH, 512);
    k_gemm_wmma<4><<<dim3(512 / 128, GG / 32), 256, 0, stream>>>(Apk, Bpk, bf1, y1, 512);
    hipMemsetAsync(csum, 0, 512 * 4, stream);
    hipMemsetAsync(csq,  0, 512 * 4, stream);
    k_bnstats<<<dim3(512, CDIV(GG, 256)), 256, 0, stream>>>(y1, csum, csq, GG, 512);
    k_bnapply<<<CDIV((size_t)GG * 512, 256), 256, 0, stream>>>(y1, csum, csq, gf1, bef1, y1, GG, 512);

    k_packA<<<CDIV((size_t)GG * 512, 256), 256, 0, stream>>>(y1, nullptr, Apk, GG, 512, 512);
    k_packB<<<CDIV((size_t)512 * 256, 256), 256, 0, stream>>>(Wl, Bpk, 512, 512, 256);
    k_gemm_wmma<16><<<dim3(256 / 128, GG / 32), 256, 0, stream>>>(Apk, Bpk, bl, y2, 256);
    hipMemsetAsync(csum, 0, 256 * 4, stream);
    hipMemsetAsync(csq,  0, 256 * 4, stream);
    k_bnstats<<<dim3(256, CDIV(GG, 256)), 256, 0, stream>>>(y2, csum, csq, GG, 256);
    k_bnapply<<<CDIV((size_t)GG * 256, 256), 256, 0, stream>>>(y2, csum, csq, gl, bel, y2, GG, 256);

    k_final<<<GG, 96, 0, stream>>>(y2, Wf2, bf2, y_out);
}
